// AutoCorrelation_53257594470781
// MI455X (gfx1250) — compile-verified, matrix operand order checked by reading, not run on hardware
//
#include <hip/hip_runtime.h>
#include <hip/hip_bf16.h>
#include <math.h>

typedef __attribute__((ext_vector_type(16))) _Float16 v16h;
typedef __attribute__((ext_vector_type(8)))  _Float16 v8h;
typedef __attribute__((ext_vector_type(8)))  float    v8f;
typedef __attribute__((ext_vector_type(4)))  int      v4i;

#if __has_builtin(__builtin_amdgcn_global_load_async_to_lds_b128)
#define HAVE_ASYNC_LDS 1
typedef __attribute__((address_space(1))) v4i gas_v4i;   // global
typedef __attribute__((address_space(3))) v4i las_v4i;   // LDS
#else
#define HAVE_ASYNC_LDS 0
#endif

__device__ inline void wait_async0() {
#if __has_builtin(__builtin_amdgcn_s_wait_asynccnt)
    __builtin_amdgcn_s_wait_asynccnt(0);
#else
    asm volatile("s_wait_asynccnt 0" ::: "memory");
#endif
}

// ---------------------------------------------------------------------------
// Conversions
// ---------------------------------------------------------------------------
__global__ void k_f32_to_f16(const float* __restrict__ in, _Float16* __restrict__ out, size_t n) {
    size_t i = (size_t)blockIdx.x * blockDim.x + threadIdx.x;
    size_t stride = (size_t)gridDim.x * blockDim.x;
    for (; i < n; i += stride) out[i] = (_Float16)in[i];
}

// out[c][r] = (f16) in[r][c]   (rows x cols f32 -> cols x rows f16)
__global__ void k_transpose_f32_to_f16(const float* __restrict__ in, _Float16* __restrict__ out,
                                       int rows, int cols) {
    int c  = blockIdx.x * 32 + threadIdx.x;
    int r0 = blockIdx.y * 32;
    if (c >= cols) return;
    #pragma unroll 4
    for (int i = 0; i < 32; ++i) {
        int r = r0 + i;
        if (r < rows) out[(size_t)c * rows + r] = (_Float16)in[(size_t)r * cols + c];
    }
}

// ---------------------------------------------------------------------------
// Column sums over time:  part[b][ch][d] = sum_{t in chunk} x[b][t][d]
// ---------------------------------------------------------------------------
__global__ void k_colsum_partial(const float* __restrict__ x, float* __restrict__ part,
                                 int L, int D, int tchunk) {
    int d  = blockIdx.x * blockDim.x + threadIdx.x;
    int ch = blockIdx.y;
    int b  = blockIdx.z;
    const float* p = x + ((size_t)b * L + (size_t)ch * tchunk) * D + d;
    float s = 0.0f;
    for (int t = 0; t < tchunk; ++t) s += p[(size_t)t * D];
    part[((size_t)b * gridDim.y + ch) * D + d] = s;
}

__global__ void k_colsum_reduce(const float* __restrict__ part, float* __restrict__ out,
                                int D, int nch) {
    int d = blockIdx.x * blockDim.x + threadIdx.x;
    int b = blockIdx.y;
    float s = 0.0f;
    for (int c = 0; c < nch; ++c) s += part[((size_t)b * nch + c) * D + d];
    out[(size_t)b * D + d] = s;
}

// ---------------------------------------------------------------------------
// Batched mat-vec: out[bb][row] = dot(W[row,:], X[bb,:]) + bias_scale*bias[row]
// one wave (32 lanes) per output row
// ---------------------------------------------------------------------------
__global__ __launch_bounds__(256)
void k_matvec(const float* __restrict__ W, const float* __restrict__ X,
              const float* __restrict__ bias, float bias_scale,
              float* __restrict__ out, int n, int kk) {
    int lane = threadIdx.x & 31;
    int w    = threadIdx.x >> 5;
    int row  = blockIdx.x * 8 + w;
    int bb   = blockIdx.y;
    if (row >= n) return;
    const float* wp = W + (size_t)row * kk;
    const float* xp = X + (size_t)bb * kk;
    float s = 0.0f;
    for (int e = lane; e < kk; e += 32) s += wp[e] * xp[e];
    #pragma unroll
    for (int off = 16; off > 0; off >>= 1) s += __shfl_xor(s, off, 32);
    if (lane == 0) {
        float bterm = bias ? bias_scale * bias[row] : 0.0f;
        out[(size_t)bb * n + row] = s + bterm;
    }
}

// ---------------------------------------------------------------------------
// mean_value, top-k selection (tie-break by lower index, like lax.top_k),
// per-batch softmax weights over the selected delays. Single 256-thread block.
// ---------------------------------------------------------------------------
__global__ __launch_bounds__(256)
void k_weights(const float* __restrict__ sQ, const float* __restrict__ sK,
               float* __restrict__ wsel /*B*64*/, int* __restrict__ sel_idx /*64*/,
               int H, int dk, int Bn, int Ltot, int ktop) {
    __shared__ float mval[4][64];
    __shared__ float mab[64];
    __shared__ int   sflag[64];
    __shared__ int   slist[64];
    __shared__ int   scount;
    int tid = threadIdx.x;
    int b = tid >> 6, c = tid & 63;
    if (b < Bn) {
        float s = 0.0f;
        int D = H * dk;
        for (int h = 0; h < H; ++h) {
            float a = sQ[(size_t)b * D + h * dk + c];
            float e = sK[(size_t)b * D + h * dk + c];
            s += a * e;
        }
        mval[b][c] = s / (float)H / (float)Ltot;
    }
    __syncthreads();
    if (tid < 64) {
        float s = 0.0f;
        for (int b2 = 0; b2 < Bn; ++b2) s += mval[b2][tid];
        mab[tid] = s / (float)Bn;
    }
    __syncthreads();
    if (tid < 64) {
        float v = mab[tid];
        int rank = 0;
        for (int c2 = 0; c2 < 64; ++c2) {
            float v2 = mab[c2];
            if (v2 > v || (v2 == v && c2 < tid)) rank++;
        }
        sflag[tid] = (rank < ktop) ? 1 : 0;
    }
    __syncthreads();
    if (tid == 0) {
        int cnt = 0;
        for (int c2 = 0; c2 < 64; ++c2) if (sflag[c2]) slist[cnt++] = c2;
        scount = cnt;
        for (int c2 = 0; c2 < 64; ++c2) sel_idx[c2] = (c2 < cnt) ? slist[c2] : 0;
    }
    __syncthreads();
    if (tid < Bn) {
        int bb = tid, cnt = scount;
        float mx = -3.4e38f;
        for (int j = 0; j < cnt; ++j) mx = fmaxf(mx, mval[bb][slist[j]]);
        float den = 0.0f;
        for (int j = 0; j < cnt; ++j) den += __expf(mval[bb][slist[j]] - mx);
        for (int j = 0; j < cnt; ++j)
            wsel[bb * 64 + j] = __expf(mval[bb][slist[j]] - mx) / den;
    }
}

// ---------------------------------------------------------------------------
// WMMA NT GEMM with double-buffered async LDS staging:
//   C[m,n] = sum_k A[m,k] * Bt[n,k]   (A: MxK f16, Bt: NxK f16, K-contig)
// Block = 256 threads = 8 waves. Block tile 128x128; wave tile 32x64.
// K-slice (128x32 halves for each of A,B) staged in LDS, padded to 40-half
// rows (80B stride -> 4-bank-spaced row starts, conflict-free b128 reads).
// Slice k+1 is copied with GLOBAL_LOAD_ASYNC_TO_LDS_B128 (ASYNCcnt) while
// slice k feeds the WMMAs; s_wait_asynccnt 0 + barrier between phases.
// ---------------------------------------------------------------------------
#define LDS_ROW 40

__global__ __launch_bounds__(256)
void k_gemm_nt_wmma(const _Float16* __restrict__ A, const _Float16* __restrict__ Bt,
                    float* __restrict__ C, int M, int N, int K) {
    __shared__ _Float16 smem[2][2][128 * LDS_ROW];   // [buf][mat][row*LDS_ROW]

    const int tid  = threadIdx.x;
    const int lane = tid & 31;
    const int wave = tid >> 5;
    const int wm   = wave >> 1;                 // 0..3
    const int wn   = wave & 1;                  // 0..1
    const int half = lane >> 4;                 // 0/1
    const int r    = lane & 15;

    // copy assignment: thread -> one 64B row (4 x b128)
    const int cmat = tid >> 7;                  // 0 = A slice, 1 = B slice
    const int crow = tid & 127;
    const _Float16* gsrc = (cmat ? (Bt + (size_t)(blockIdx.x * 128 + crow) * K)
                                 : (A  + (size_t)(blockIdx.y * 128 + crow) * K));

    auto issue_copy = [&](int buf, int k0) {
        const _Float16* g = gsrc + k0;
        _Float16* l = &smem[buf][cmat][crow * LDS_ROW];
#if HAVE_ASYNC_LDS
        #pragma unroll
        for (int c = 0; c < 4; ++c)
            __builtin_amdgcn_global_load_async_to_lds_b128(
                (gas_v4i*)(void*)(g + c * 8),
                (las_v4i*)(void*)(l + c * 8), 0, 0);
#else
        #pragma unroll
        for (int c = 0; c < 4; ++c)
            *(v8h*)(l + c * 8) = *(const v8h*)(g + c * 8);
#endif
    };

    auto load_frag = [&](int buf, int mat, int rbase) -> v16h {
        const _Float16* p = &smem[buf][mat][(rbase + r) * LDS_ROW + 8 * half];
        v8h lo = *(const v8h*)p;
        v8h hi = *(const v8h*)(p + 16);
        v16h o;
        #pragma unroll
        for (int i = 0; i < 8; ++i) { o[i] = lo[i]; o[i + 8] = hi[i]; }
        return o;
    };

    v8f acc[2][4];
    #pragma unroll
    for (int i = 0; i < 2; ++i)
        #pragma unroll
        for (int j = 0; j < 4; ++j)
            #pragma unroll
            for (int e = 0; e < 8; ++e) acc[i][j][e] = 0.0f;

    const int NIT = K / 32;
    issue_copy(0, 0);

    for (int it = 0; it < NIT; ++it) {
        const int cur = it & 1;
#if HAVE_ASYNC_LDS
        wait_async0();                    // own async copies of `cur` landed
#endif
        __syncthreads();                  // everyone's copies landed; prev reads done
        if (it + 1 < NIT) issue_copy(1 - cur, (it + 1) * 32);

        v16h a[2], b[4];
        #pragma unroll
        for (int i = 0; i < 2; ++i) a[i] = load_frag(cur, 0, wm * 32 + i * 16);
        #pragma unroll
        for (int j = 0; j < 4; ++j) b[j] = load_frag(cur, 1, wn * 64 + j * 16);
        #pragma unroll
        for (int i = 0; i < 2; ++i)
            #pragma unroll
            for (int j = 0; j < 4; ++j)
                acc[i][j] = __builtin_amdgcn_wmma_f32_16x16x32_f16(
                    false, a[i], false, b[j], (short)0, acc[i][j], false, false);
    }

    // C/D layout: VGPR e -> row m0+i*16+8*half+e, col n0+j*16+r
    const int m0 = blockIdx.y * 128 + wm * 32;
    const int n0 = blockIdx.x * 128 + wn * 64;
    #pragma unroll
    for (int i = 0; i < 2; ++i)
        #pragma unroll
        for (int j = 0; j < 4; ++j) {
            float* cp = C + (size_t)(m0 + i * 16 + 8 * half) * N + n0 + j * 16 + r;
            #pragma unroll
            for (int e = 0; e < 8; ++e) cp[(size_t)e * N] = acc[i][j][e];
        }
}

// ---------------------------------------------------------------------------
// out[b][t][d] = sum_j w[b][j] * VO[b][(t+sel[j])%L][d] + bias[d]
// LDS tile: (64+63) rows x 64 cols of f32
// ---------------------------------------------------------------------------
#define T_TILE 64
#define D_TILE 64
__global__ __launch_bounds__(256)
void k_agg_out(const float* __restrict__ VO, const float* __restrict__ wsel,
               const int* __restrict__ sel_idx, const float* __restrict__ bias,
               float* __restrict__ out, int L, int D, int nsel) {
    __shared__ float tile[T_TILE + 63][D_TILE];
    __shared__ float wsh[64];
    __shared__ int   ish[64];
    int b  = blockIdx.z;
    int t0 = blockIdx.y * T_TILE;
    int d0 = blockIdx.x * D_TILE;

    const int total = (T_TILE + 63) * D_TILE;
    for (int i = threadIdx.x; i < total; i += 256) {
        int rr = i / D_TILE, dd = i % D_TILE;
        int t = t0 + rr; if (t >= L) t -= L;             // circular wrap
        tile[rr][dd] = VO[((size_t)b * L + t) * D + d0 + dd];
    }
    if (threadIdx.x < 64) {
        wsh[threadIdx.x] = wsel[b * 64 + threadIdx.x];
        ish[threadIdx.x] = sel_idx[threadIdx.x];
    }
    __syncthreads();

    int dd  = threadIdx.x & 63;
    int tl0 = threadIdx.x >> 6;                           // 0..3
    float bd = bias[d0 + dd];
    for (int tl = tl0; tl < T_TILE; tl += 4) {
        float s = bd;
        for (int j = 0; j < nsel; ++j) s += wsh[j] * tile[tl + ish[j]][dd];
        out[((size_t)b * L + t0 + tl) * D + d0 + dd] = s;
    }
}

// ---------------------------------------------------------------------------
// Host-side orchestration
// ---------------------------------------------------------------------------
extern "C" void kernel_launch(void* const* d_in, const int* in_sizes, int n_in,
                              void* d_out, int out_size, void* d_ws, size_t ws_size,
                              hipStream_t stream) {
    const float* q  = (const float*)d_in[0];
    const float* k  = (const float*)d_in[1];
    const float* v  = (const float*)d_in[2];
    const float* Wq = (const float*)d_in[3];
    const float* bq = (const float*)d_in[4];
    const float* Wk = (const float*)d_in[5];
    const float* bk = (const float*)d_in[6];
    const float* Wv = (const float*)d_in[7];
    const float* bv = (const float*)d_in[8];
    const float* Wo = (const float*)d_in[9];
    const float* bo = (const float*)d_in[10];
    float* out = (float*)d_out;

    const int B = 4, L = 4096, D = 1024, H = 16, dk = 64;
    const int NCH = 8, TCH = L / NCH;
    int ktop = (int)(5.0 * log((double)(L + 1)));
    if (ktop < 1) ktop = 1;
    if (ktop > L) ktop = L;
    if (ktop > dk) ktop = dk;                 // actual_k = min(top_k, C)

    // workspace carve-out (256B aligned)
    char* p = (char*)d_ws;
    auto alloc = [&](size_t bytes) -> void* {
        void* r = (void*)p;
        p += (bytes + 255) & ~(size_t)255;
        return r;
    };
    float*    part_q = (float*)alloc((size_t)B * NCH * D * 4);
    float*    part_k = (float*)alloc((size_t)B * NCH * D * 4);
    float*    sq     = (float*)alloc((size_t)B * D * 4);
    float*    sk     = (float*)alloc((size_t)B * D * 4);
    float*    sQ     = (float*)alloc((size_t)B * D * 4);
    float*    sK     = (float*)alloc((size_t)B * D * 4);
    float*    biasT  = (float*)alloc((size_t)D * 4);          // Wo@bv + bo
    float*    wsel   = (float*)alloc((size_t)B * 64 * 4);
    int*      selidx = (int*)alloc(64 * 4);
    float*    Wf32   = (float*)alloc((size_t)D * D * 4);      // Wo@Wv
    _Float16* Woh    = (_Float16*)alloc((size_t)D * D * 2);
    _Float16* WvTh   = (_Float16*)alloc((size_t)D * D * 2);
    _Float16* Wh     = (_Float16*)alloc((size_t)D * D * 2);
    _Float16* vh     = (_Float16*)alloc((size_t)B * L * D * 2);
    float*    VO     = (float*)alloc((size_t)B * L * D * 4);

    // 1) f16 conversions
    {
        size_t n = (size_t)B * L * D;
        k_f32_to_f16<<<dim3((unsigned)((n + 255) / 256)), dim3(256), 0, stream>>>(v, vh, n);
        size_t m = (size_t)D * D;
        k_f32_to_f16<<<dim3((unsigned)((m + 255) / 256)), dim3(256), 0, stream>>>(Wo, Woh, m);
        k_transpose_f32_to_f16<<<dim3(D / 32, D / 32), dim3(32), 0, stream>>>(Wv, WvTh, D, D);
    }

    // 2) column sums of q, k over time
    k_colsum_partial<<<dim3(D / 256, NCH, B), dim3(256), 0, stream>>>(q, part_q, L, D, TCH);
    k_colsum_partial<<<dim3(D / 256, NCH, B), dim3(256), 0, stream>>>(k, part_k, L, D, TCH);
    k_colsum_reduce<<<dim3(D / 256, B), dim3(256), 0, stream>>>(part_q, sq, D, NCH);
    k_colsum_reduce<<<dim3(D / 256, B), dim3(256), 0, stream>>>(part_k, sk, D, NCH);

    // 3) sQ[b] = Wq@sq[b] + L*bq ; sK[b] = Wk@sk[b] + L*bk ; biasT = Wo@bv + bo
    k_matvec<<<dim3(D / 8, B), dim3(256), 0, stream>>>(Wq, sq, bq, (float)L, sQ, D, D);
    k_matvec<<<dim3(D / 8, B), dim3(256), 0, stream>>>(Wk, sk, bk, (float)L, sK, D, D);
    k_matvec<<<dim3(D / 8, 1), dim3(256), 0, stream>>>(Wo, bv, bo, 1.0f, biasT, D, D);

    // 4) delay selection + softmax weights
    k_weights<<<dim3(1), dim3(256), 0, stream>>>(sQ, sK, wsel, selidx, H, dk, B, L, ktop);

    // 5) W = Wo @ Wv  (NT via WvT), f32 accumulate -> cast to f16
    k_gemm_nt_wmma<<<dim3(D / 128, D / 128), dim3(256), 0, stream>>>(Woh, WvTh, Wf32, D, D, D);
    {
        size_t m = (size_t)D * D;
        k_f32_to_f16<<<dim3((unsigned)((m + 255) / 256)), dim3(256), 0, stream>>>(Wf32, Wh, m);
    }

    // 6) VO = v @ W^T  (the single big GEMM, 16384x1024x1024)
    k_gemm_nt_wmma<<<dim3(D / 128, (B * L) / 128), dim3(256), 0, stream>>>(vh, Wh, VO, B * L, D, D);

    // 7) 41-tap circular delay aggregation + bias -> out
    k_agg_out<<<dim3(D / D_TILE, L / T_TILE, B), dim3(256), 0, stream>>>(
        VO, wsel, selidx, biasT, out, L, D, ktop);
}
